// ConditionalAttentionLayer_36696200577205
// MI455X (gfx1250) — compile-verified, matrix operand order checked by reading, not run on hardware
//
#include <hip/hip_runtime.h>
#include <hip/hip_bf16.h>

// ---------------------------------------------------------------------------
// Problem constants (Cora conditional-GAT forward)
// ---------------------------------------------------------------------------
#define NN     2708            // nodes
#define INS    1433            // input features
#define NMECH  8               // mechanisms
#define OUTS   64              // per-mechanism output width
#define COLS   576             // fused GEMM cols: 8*64 proj + 64 conditioner hidden
#define KPAD   1440            // INS padded to multiple of 32
#define NPAD   2752            // NN padded to multiple of 64 (GEMM rows)
#define JPAD   2720            // NN padded to multiple of 32 (attention j dim)
#define JMAIN  2688            // largest multiple of 32 where j0+8 <= NN always holds
#define LEAK   0.2f

typedef __attribute__((ext_vector_type(16))) _Float16 v16h;
typedef __attribute__((ext_vector_type(8)))  float    v8f;

union HFrag {
    uint4    q[2];   // two 16-byte chunks
    _Float16 h[16];
    v16h     v;
};

// ---------------------------------------------------------------------------
// Kernel 1: pack x -> f16, zero-padded [NPAD][KPAD]
// ---------------------------------------------------------------------------
__global__ __launch_bounds__(256) void pack_x_kernel(const float* __restrict__ x,
                                                     _Float16* __restrict__ Xh) {
    size_t idx = (size_t)blockIdx.x * 256 + threadIdx.x;
    if (idx >= (size_t)NPAD * KPAD) return;
    int r = (int)(idx / KPAD);
    int k = (int)(idx % KPAD);
    float v = (r < NN && k < INS) ? x[(size_t)r * INS + k] : 0.f;
    Xh[idx] = (_Float16)v;
}

// ---------------------------------------------------------------------------
// Kernel 2: pack fused weights [W(m,f,o)->cols 0..511 | Wc1->cols 512..575]
//           transposed to [COLS][KPAD] f16 so B fragments load contiguously.
// ---------------------------------------------------------------------------
__global__ __launch_bounds__(256) void pack_B_kernel(const float* __restrict__ W,
                                                     const float* __restrict__ Wc1,
                                                     _Float16* __restrict__ Bt) {
    size_t idx = (size_t)blockIdx.x * 256 + threadIdx.x;
    if (idx >= (size_t)COLS * KPAD) return;
    int c = (int)(idx / KPAD);
    int k = (int)(idx % KPAD);
    float v = 0.f;
    if (k < INS) {
        if (c < NMECH * OUTS) {
            int mm = c >> 6, o = c & 63;
            v = W[((size_t)mm * INS + k) * OUTS + o];
        } else {
            v = Wc1[(size_t)k * 64 + (c - NMECH * OUTS)];
        }
    }
    Bt[idx] = (_Float16)v;
}

// ---------------------------------------------------------------------------
// Kernel 3: C[NPAD][COLS] = Xh[NPAD][KPAD] @ B  via v_wmma_f32_16x16x32_f16.
// 64x64 block tile, 8 waves; each wave owns a 16x32 output subtile.
//
// Double-buffered CDNA5 async global->LDS pipeline:
//   - tile k+1 issued with global_load_async_to_lds_b128 (ASYNCcnt) while
//     tile k is consumed by WMMA;
//   - async loads complete IN ORDER (ISA 15.5/§7.1), so s_wait_asynccnt 2
//     guarantees the older tile pair has landed while the newer stays in
//     flight; buffer reuse is fenced by the end-of-iteration barrier.
// ---------------------------------------------------------------------------
__global__ __launch_bounds__(256) void gemm_f16_wmma(const _Float16* __restrict__ A,
                                                     const _Float16* __restrict__ B,
                                                     float* __restrict__ C) {
    __shared__ __align__(16) _Float16 sA[2][64][32];   // 2 x 4 KB
    __shared__ __align__(16) _Float16 sB[2][64][32];   // 2 x 4 KB

    const int tid  = threadIdx.x;
    const int wave = tid >> 5;
    const int lane = tid & 31;
    const int rowBase = blockIdx.x * 64;
    const int colBase = blockIdx.y * 64;
    const int wr = (wave >> 1) * 16;   // wave's row offset within block tile
    const int wc = (wave & 1) * 32;    // wave's col offset within block tile
    const int hs  = lane >> 4;         // half-select per ISA 16-bit operand layout
    const int l16 = lane & 15;

    const int ldRow = tid >> 2;        // 0..63
    const int ldK   = (tid & 3) * 8;   // 0,8,16,24

    // per-lane LDS byte addresses (flat LDS aperture: low 32 bits = LDS offset)
    const unsigned ldsA0 = (unsigned)(size_t)&sA[0][ldRow][ldK];
    const unsigned ldsB0 = (unsigned)(size_t)&sB[0][ldRow][ldK];
    const _Float16* gA = A + (size_t)(rowBase + ldRow) * KPAD + ldK;
    const _Float16* gB = B + (size_t)(colBase + ldRow) * KPAD + ldK;

    auto issue_tile = [&](int ktile, int buf) {
        unsigned long long ga = (unsigned long long)(size_t)(gA + ktile);
        unsigned long long gb = (unsigned long long)(size_t)(gB + ktile);
        unsigned la = ldsA0 + (unsigned)buf * (unsigned)sizeof(sA[0]);
        unsigned lb = ldsB0 + (unsigned)buf * (unsigned)sizeof(sB[0]);
        asm volatile("global_load_async_to_lds_b128 %0, %1, off"
                     :: "v"(la), "v"(ga) : "memory");
        asm volatile("global_load_async_to_lds_b128 %0, %1, off"
                     :: "v"(lb), "v"(gb) : "memory");
    };

    v8f acc0 = {};
    v8f acc1 = {};

    issue_tile(0, 0);                         // prologue: tile 0 in flight
    int ib = 0;
    for (int kt = 0; kt < KPAD; kt += 32) {
        if (kt + 32 < KPAD) {
            issue_tile(kt + 32, ib ^ 1);      // next tile overlaps this compute
            asm volatile("s_wait_asynccnt 0x2" ::: "memory");   // oldest pair done
        } else {
            asm volatile("s_wait_asynccnt 0x0" ::: "memory");
        }
        if (kt + 64 < KPAD) {                 // L2 prefetch two tiles ahead
            __builtin_prefetch(gA + kt + 64, 0, 1);
            __builtin_prefetch(gB + kt + 64, 0, 1);
        }
        __syncthreads();                      // all waves' tile data visible

        const _Float16 (*tA)[32] = sA[ib];
        const _Float16 (*tB)[32] = sB[ib];
        HFrag a, b0, b1;
        a.q[0]  = *(const uint4*)&tA[wr + l16][hs * 8];
        a.q[1]  = *(const uint4*)&tA[wr + l16][16 + hs * 8];
        b0.q[0] = *(const uint4*)&tB[wc + l16][hs * 8];
        b0.q[1] = *(const uint4*)&tB[wc + l16][16 + hs * 8];
        b1.q[0] = *(const uint4*)&tB[wc + 16 + l16][hs * 8];
        b1.q[1] = *(const uint4*)&tB[wc + 16 + l16][16 + hs * 8];

        acc0 = __builtin_amdgcn_wmma_f32_16x16x32_f16(false, a.v, false, b0.v,
                                                      (short)0, acc0, false, false);
        acc1 = __builtin_amdgcn_wmma_f32_16x16x32_f16(false, a.v, false, b1.v,
                                                      (short)0, acc1, false, false);
        __syncthreads();                      // fences buffer reuse at next issue
        ib ^= 1;
    }

    // C/D layout: VGPR i -> row M=i (lanes 0-15) / M=8+i (lanes 16-31), col=lane%16
#pragma unroll
    for (int i = 0; i < 8; ++i) {
        int r = rowBase + wr + i + hs * 8;
        C[(size_t)r * COLS + colBase + wc + l16]      = acc0[i];
        C[(size_t)r * COLS + colBase + wc + 16 + l16] = acc1[i];
    }
}

// ---------------------------------------------------------------------------
// Kernel 4: conditioner tail + FiLM + attention projections.
// One 64-thread block per (padded) node.
//   hidden = relu(C[:,512:576]+bc1); cond = hidden@Wc2+bc2
//   h[m,n,o] = gamma*h + beta  -> Ht f16 [m*64+o][JPAD] (transposed for P@V)
//   esrc/edst[m,n] = dot(h, a_src/a_dst)
// ---------------------------------------------------------------------------
__global__ __launch_bounds__(64) void cond_film_kernel(
        const float* __restrict__ Cmat,
        const float* __restrict__ bc1, const float* __restrict__ Wc2,
        const float* __restrict__ bc2,
        const float* __restrict__ a_src, const float* __restrict__ a_dst,
        _Float16* __restrict__ Ht,
        float* __restrict__ esrc, float* __restrict__ edst) {
    __shared__ float shid[64];
    __shared__ float scond[16];
    __shared__ float sh[64];
    const int n = blockIdx.x;
    const int t = threadIdx.x;

    if (n >= NN) {                 // padding rows: zero everything consumers read
        if (n < JPAD) {
            for (int m = 0; m < NMECH; ++m)
                Ht[(size_t)(m * 64 + t) * JPAD + n] = (_Float16)0.f;
        }
        if (t < NMECH) {
            esrc[t * NPAD + n] = 0.f;
            edst[t * NPAD + n] = 0.f;
        }
        return;
    }

    float hv = Cmat[(size_t)n * COLS + 512 + t] + bc1[t];
    shid[t] = hv > 0.f ? hv : 0.f;
    __syncthreads();

    if (t < 16) {
        float s = bc2[t];
        for (int o = 0; o < 64; ++o) s += shid[o] * Wc2[o * 16 + t];
        scond[t] = s;
    }
    __syncthreads();

    for (int m = 0; m < NMECH; ++m) {
        float g = scond[m], b = scond[8 + m];
        float hval = g * Cmat[(size_t)n * COLS + m * 64 + t] + b;
        sh[t] = hval;
        Ht[(size_t)(m * 64 + t) * JPAD + n] = (_Float16)hval;
        __syncthreads();
        if (t < 2) {
            const float* av = (t == 0 ? a_src : a_dst) + m * 64;
            float s = 0.f;
            for (int o = 0; o < 64; ++o) s += sh[o] * av[o];
            (t == 0 ? esrc : edst)[m * NPAD + n] = s;
        }
        __syncthreads();
    }
}

// ---------------------------------------------------------------------------
// Kernel 5: per-(m,row) masked softmax stats (max and 1/sumexp).
// adj rows are exactly NN/4 = 677 int4, edst is float4-aligned -> all b128.
// ---------------------------------------------------------------------------
__global__ __launch_bounds__(256) void attn_stats_kernel(
        const int* __restrict__ adj,
        const float* __restrict__ esrc, const float* __restrict__ edst,
        float* __restrict__ rowmax, float* __restrict__ rowinv) {
    __shared__ float red[256];
    const int i = blockIdx.x;
    const int m = blockIdx.y;
    const int t = threadIdx.x;
    if (i >= NN) {
        if (t == 0) { rowmax[m * NPAD + i] = 0.f; rowinv[m * NPAD + i] = 0.f; }
        return;
    }
    const float es = esrc[m * NPAD + i];
    const int4*   adjrow = (const int4*)(adj + (size_t)i * NN);
    const float4* ed4    = (const float4*)(edst + m * NPAD);
    const int nq = NN >> 2;   // 677, exact (NN % 4 == 0)

    float mx = -1e30f;
#pragma unroll
    for (int it = 0; it < 3; ++it) {
        int q = t + it * 256;
        if (q < nq) {
            int4   a = adjrow[q];
            float4 e = ed4[q];
            if (a.x) { float s = es + e.x; s = s > 0.f ? s : LEAK * s; mx = fmaxf(mx, s); }
            if (a.y) { float s = es + e.y; s = s > 0.f ? s : LEAK * s; mx = fmaxf(mx, s); }
            if (a.z) { float s = es + e.z; s = s > 0.f ? s : LEAK * s; mx = fmaxf(mx, s); }
            if (a.w) { float s = es + e.w; s = s > 0.f ? s : LEAK * s; mx = fmaxf(mx, s); }
        }
    }
    red[t] = mx; __syncthreads();
    for (int w = 128; w > 0; w >>= 1) {
        if (t < w) red[t] = fmaxf(red[t], red[t + w]);
        __syncthreads();
    }
    mx = red[0]; __syncthreads();

    float sum = 0.f;
#pragma unroll
    for (int it = 0; it < 3; ++it) {
        int q = t + it * 256;
        if (q < nq) {
            int4   a = adjrow[q];
            float4 e = ed4[q];
            if (a.x) { float s = es + e.x; s = s > 0.f ? s : LEAK * s; sum += __expf(s - mx); }
            if (a.y) { float s = es + e.y; s = s > 0.f ? s : LEAK * s; sum += __expf(s - mx); }
            if (a.z) { float s = es + e.z; s = s > 0.f ? s : LEAK * s; sum += __expf(s - mx); }
            if (a.w) { float s = es + e.w; s = s > 0.f ? s : LEAK * s; sum += __expf(s - mx); }
        }
    }
    red[t] = sum; __syncthreads();
    for (int w = 128; w > 0; w >>= 1) {
        if (t < w) red[t] += red[t + w];
        __syncthreads();
    }
    if (t == 0) { rowmax[m * NPAD + i] = mx; rowinv[m * NPAD + i] = 1.f / red[0]; }
}

// ---------------------------------------------------------------------------
// Kernel 6: out[i, m*64+o] = sum_j P[m,i,j] * h[m,j,o]  via WMMA.
// 4 waves/block, each wave owns one (m, 16-row) tile (172*4 = 688 tiles; rows >= NN masked).
// P is built in registers as the 16x32 f16 A-fragment; adj/edst for a lane's
// 8 consecutive j are fetched as b128 vector loads on the main (in-bounds)
// chunks, with a guarded scalar tail for j in [2688, 2720).
// ---------------------------------------------------------------------------
__global__ __launch_bounds__(128) void attn_av_wmma(
        const int* __restrict__ adj,
        const float* __restrict__ esrc, const float* __restrict__ edst,
        const float* __restrict__ rowmax, const float* __restrict__ rowinv,
        const _Float16* __restrict__ Ht,
        float* __restrict__ out) {
    const int m     = blockIdx.y;
    const int wave  = threadIdx.x >> 5;
    const int lane  = threadIdx.x & 31;
    const int tile  = blockIdx.x * 4 + wave;   // 0..687
    const int ibase = tile * 16;
    const int hs    = lane >> 4;
    const int l16   = lane & 15;
    const int row   = ibase + l16;        // A-fragment row owned by this lane
    const bool rvalid = row < NN;

    const float es  = rvalid ? esrc[m * NPAD + row]   : 0.f;
    const float mx  = rvalid ? rowmax[m * NPAD + row] : 0.f;
    const float inv = rvalid ? rowinv[m * NPAD + row] : 0.f;

    const float* edm = edst + m * NPAD;
    const _Float16* htm = Ht + (size_t)(m * 64 + l16) * JPAD;

    v8f acc[4] = {};

    // main chunks: every lane's 8-j group is fully in-bounds -> b128 loads
    for (int jb = 0; jb < JMAIN; jb += 32) {
        HFrag pa;
#pragma unroll
        for (int g = 0; g < 2; ++g) {
            const int j0 = jb + g * 16 + hs * 8;
            int aa[8];
            float ee[8];
            if (rvalid) {
                const int4* ap4 = (const int4*)(adj + (size_t)row * NN + j0);
                int4 a0 = ap4[0], a1 = ap4[1];
                aa[0] = a0.x; aa[1] = a0.y; aa[2] = a0.z; aa[3] = a0.w;
                aa[4] = a1.x; aa[5] = a1.y; aa[6] = a1.z; aa[7] = a1.w;
            } else {
#pragma unroll
                for (int kk = 0; kk < 8; ++kk) aa[kk] = 0;
            }
            const float4* ep4 = (const float4*)(edm + j0);
            float4 e0 = ep4[0], e1 = ep4[1];
            ee[0] = e0.x; ee[1] = e0.y; ee[2] = e0.z; ee[3] = e0.w;
            ee[4] = e1.x; ee[5] = e1.y; ee[6] = e1.z; ee[7] = e1.w;
#pragma unroll
            for (int kk = 0; kk < 8; ++kk) {
                float p = 0.f;
                if (aa[kk]) {
                    float s = es + ee[kk];
                    s = s > 0.f ? s : LEAK * s;
                    p = __expf(s - mx) * inv;
                }
                pa.h[g * 8 + kk] = (_Float16)p;
            }
        }
#pragma unroll
        for (int cb = 0; cb < 4; ++cb) {
            HFrag bf;
            const _Float16* hp = htm + (size_t)(cb * 16) * JPAD + jb;
            bf.q[0] = *(const uint4*)(hp + hs * 8);
            bf.q[1] = *(const uint4*)(hp + 16 + hs * 8);
            acc[cb] = __builtin_amdgcn_wmma_f32_16x16x32_f16(false, pa.v, false, bf.v,
                                                             (short)0, acc[cb],
                                                             false, false);
        }
    }

    // guarded tail chunk jb = JMAIN (j in [2688, 2720), crosses NN)
    {
        const int jb = JMAIN;
        HFrag pa;
#pragma unroll
        for (int g = 0; g < 2; ++g) {
#pragma unroll
            for (int kk = 0; kk < 8; ++kk) {
                int j = jb + g * 16 + hs * 8 + kk;
                float p = 0.f;
                if (rvalid && j < NN && adj[(size_t)row * NN + j]) {
                    float s = es + edm[j];
                    s = s > 0.f ? s : LEAK * s;
                    p = __expf(s - mx) * inv;
                }
                pa.h[g * 8 + kk] = (_Float16)p;
            }
        }
#pragma unroll
        for (int cb = 0; cb < 4; ++cb) {
            HFrag bf;
            const _Float16* hp = htm + (size_t)(cb * 16) * JPAD + jb;
            bf.q[0] = *(const uint4*)(hp + hs * 8);
            bf.q[1] = *(const uint4*)(hp + 16 + hs * 8);
            acc[cb] = __builtin_amdgcn_wmma_f32_16x16x32_f16(false, pa.v, false, bf.v,
                                                             (short)0, acc[cb],
                                                             false, false);
        }
    }

#pragma unroll
    for (int cb = 0; cb < 4; ++cb) {
#pragma unroll
        for (int i = 0; i < 8; ++i) {
            int r = ibase + i + hs * 8;
            if (r < NN)
                out[(size_t)r * (NMECH * OUTS) + m * OUTS + cb * 16 + l16] = acc[cb][i];
        }
    }
}

// ---------------------------------------------------------------------------
// Host-side launch. Workspace layout (all 16B aligned, total ~19.1 MB):
//   Xh   f16 [NPAD][KPAD]   7,925,760 B
//   Bt   f16 [COLS][KPAD]   1,658,880 B
//   Cmat f32 [NPAD][COLS]   6,340,608 B
//   Ht   f16 [512][JPAD]    2,785,280 B
//   esrc/edst/rowmax/rowinv f32 [8][NPAD]  4 x 88,064 B
// ---------------------------------------------------------------------------
extern "C" void kernel_launch(void* const* d_in, const int* in_sizes, int n_in,
                              void* d_out, int out_size, void* d_ws, size_t ws_size,
                              hipStream_t stream) {
    (void)in_sizes; (void)n_in; (void)out_size; (void)ws_size;

    const float* x     = (const float*)d_in[0];
    const int*   adj   = (const int*)d_in[1];   // bool mask delivered as int per harness
    const float* W     = (const float*)d_in[2];
    const float* a_src = (const float*)d_in[3];
    const float* a_dst = (const float*)d_in[4];
    const float* Wc1   = (const float*)d_in[5];
    const float* bc1   = (const float*)d_in[6];
    const float* Wc2   = (const float*)d_in[7];
    const float* bc2   = (const float*)d_in[8];
    float* out = (float*)d_out;

    char* ws = (char*)d_ws;
    size_t off = 0;
    _Float16* Xh = (_Float16*)(ws + off); off += (size_t)NPAD * KPAD * 2;
    _Float16* Bt = (_Float16*)(ws + off); off += (size_t)COLS * KPAD * 2;
    float* Cmat  = (float*)(ws + off);    off += (size_t)NPAD * COLS * 4;
    _Float16* Ht = (_Float16*)(ws + off); off += (size_t)NMECH * OUTS * JPAD * 2;
    float* esrc  = (float*)(ws + off);    off += (size_t)NMECH * NPAD * 4;
    float* edst  = (float*)(ws + off);    off += (size_t)NMECH * NPAD * 4;
    float* rmax  = (float*)(ws + off);    off += (size_t)NMECH * NPAD * 4;
    float* rinv  = (float*)(ws + off);    off += (size_t)NMECH * NPAD * 4;

    // 1) pack inputs to f16
    {
        size_t tot = (size_t)NPAD * KPAD;
        pack_x_kernel<<<dim3((unsigned)((tot + 255) / 256)), dim3(256), 0, stream>>>(x, Xh);
    }
    {
        size_t tot = (size_t)COLS * KPAD;
        pack_B_kernel<<<dim3((unsigned)((tot + 255) / 256)), dim3(256), 0, stream>>>(W, Wc1, Bt);
    }
    // 2) fused projection + conditioner-hidden GEMM (WMMA, pipelined async-LDS)
    gemm_f16_wmma<<<dim3(NPAD / 64, COLS / 64), dim3(256), 0, stream>>>(Xh, Bt, Cmat);
    // 3) conditioner tail + FiLM + e_src/e_dst
    cond_film_kernel<<<dim3(NPAD), dim3(64), 0, stream>>>(Cmat, bc1, Wc2, bc2,
                                                          a_src, a_dst, Ht, esrc, edst);
    // 4) softmax stats
    attn_stats_kernel<<<dim3(NPAD, NMECH), dim3(256), 0, stream>>>(adj, esrc, edst, rmax, rinv);
    // 5) P @ V with WMMA, direct store into [N, M*OUTS]
    attn_av_wmma<<<dim3(NPAD / 64 * 16 / 4, NMECH), dim3(128), 0, stream>>>(adj, esrc, edst,
                                                                            rmax, rinv, Ht, out);
}
